// SphereVoxelConv_5007931867713
// MI455X (gfx1250) — compile-verified, matrix operand order checked by reading, not run on hardware
//
#include <hip/hip_runtime.h>
#include <stdint.h>

// ---------------------------------------------------------------------------
// DeepSphere U-Net forward on MI455X (gfx1250, wave32, WMMA bf16 + TDM).
// Activations node-major (N, B2=256, F) row-major, so every Laplacian
// multiply is a single GEMM (N x N) @ (N x B2*F).
// ---------------------------------------------------------------------------

typedef __bf16 bf16_t;
typedef __attribute__((ext_vector_type(16))) __bf16 v16bf;
typedef __attribute__((ext_vector_type(8)))  float  v8f;

#define B2 256            // effective batch = 4*8*8
#define EPSBN 1e-5f

__device__ __forceinline__ uint32_t pack2(bf16_t lo, bf16_t hi) {
  union { bf16_t b[2]; uint32_t u; } t;
  t.b[0] = lo; t.b[1] = hi;
  return t.u;
}

// ---- Tensor Data Mover availability (device pass only) --------------------
#if defined(__has_builtin)
#if __has_builtin(__builtin_amdgcn_tensor_load_to_lds) && \
    __has_builtin(__builtin_amdgcn_s_wait_tensorcnt)
#define USE_TDM 1
#endif
#endif
#ifndef USE_TDM
#define USE_TDM 0
#endif

#if USE_TDM
typedef __attribute__((ext_vector_type(4))) unsigned int tdm_u32x4;
typedef __attribute__((ext_vector_type(8))) int          tdm_i32x8;
typedef __attribute__((ext_vector_type(4))) int          tdm_i32x4;

// DMA a 2D bf16 tile (tile_dim0=32 along k, tile_dim1=128 rows) from global
// (row stride = strideElems) into LDS at lds_off. LDS rows are padded to
// 17 dwords via pad_enable (interval 16 dwords, amount 1 dword). Rows >=
// tensRows and k >= tensK are zero-filled by TDM OOB handling.
__device__ __forceinline__ void tdm_load_tile_A(const bf16_t* g, uint32_t lds_off,
                                                uint32_t tensK, uint32_t tensRows,
                                                uint32_t strideElems)
{
  uint64_t ga = (uint64_t)(uintptr_t)g;
  tdm_u32x4 g0;
  g0[0] = 1u;                                              // count=1 (user D#)
  g0[1] = lds_off;                                         // LDS byte address
  g0[2] = (uint32_t)ga;                                    // global addr lo
  g0[3] = (uint32_t)((ga >> 32) & 0x01FFFFFFu) | (2u << 30); // addr hi | type=2
  tdm_i32x8 g1;
  g1[0] = (int)((1u << 16) | (1u << 20) | (3u << 22));     // 2B elems, pad 16dw+1dw
  g1[1] = (int)((tensK & 0xFFFFu) << 16);                  // tensor_dim0 lo16
  g1[2] = (int)(((tensK >> 16) & 0xFFFFu) | ((tensRows & 0xFFFFu) << 16));
  g1[3] = (int)(((tensRows >> 16) & 0xFFFFu) | (32u << 16)); // tile_dim0 = 32
  g1[4] = (int)128;                                        // tile_dim1=128, dim2=0
  g1[5] = (int)strideElems;                                // dim0 stride lo32
  g1[6] = 0;
  g1[7] = 0;
  tdm_i32x4 z4 = {0, 0, 0, 0};
#if __clang_major__ >= 23
  tdm_i32x8 z8 = {0, 0, 0, 0, 0, 0, 0, 0};
  __builtin_amdgcn_tensor_load_to_lds(g0, g1, z4, z4, z8, 0);
#else
  __builtin_amdgcn_tensor_load_to_lds(g0, g1, z4, z4, 0);
#endif
}
#endif // USE_TDM

// ---------------------------------------------------------------------------
// General bf16 WMMA GEMM:
//   C = alpha * ( sum_{t<nTerms} A_t @ B_t ) + beta * Addend
// A_t: (M x Kd) bf16 row-major, lda, term stride sAT (elements)
// B_t: (Kd x N) bf16 row-major, ldb, term stride sBT (elements)
// Output: fp32 (Cf) or bf16 (Cbf).
// Block tile 128x64, K-step 32, 8 wave32s (each wave 32x32 = 2x2 WMMA frags).
// A tile staged by TDM (double buffered, wave 0 issues + s_wait_tensorcnt);
// B tile staged manually with an in-register k-pair transpose.
// Term/k coordinates tracked with running counters (no integer division).
// ---------------------------------------------------------------------------
__global__ __launch_bounds__(256)
void k_gemm_wmma(const bf16_t* __restrict__ A, long long lda, long long sAT,
                 const bf16_t* __restrict__ B, long long ldb, long long sBT,
                 int nTerms, long long M, long long N, int Kd,
                 float alpha, const bf16_t* __restrict__ Add, long long ldadd,
                 float beta,
                 float* __restrict__ Cf, bf16_t* __restrict__ Cbf,
                 long long ldc)
{
  __shared__ uint32_t As[2][128 * 17];  // [buf][row][kpair], +1 dword row pad
  __shared__ uint32_t Bs[64 * 17];      // [col][kpair] (transposed at stage)

  const int tid  = threadIdx.x;
  const int lane = tid & 31;
  const int wave = tid >> 5;            // 0..7
  const int wm   = wave >> 1;           // 0..3  row group (32 rows each)
  const int wn   = wave & 1;            // 0..1  col group (32 cols each)
  const long long bm = (long long)blockIdx.y * 128;
  const long long bn = (long long)blockIdx.x * 64;
  const int half16 = (lane >> 4) & 1;   // lanes 16..31 hold K=8..15 / 24..31
  const int rc     = lane & 15;

  // B staging footprint: 2 k x 4 n
  const int bkp  = tid >> 4;            // k-pair 0..15
  const int bn4  = (tid & 15) << 2;     // n offset 0..60
  const int bk   = bkp << 1;            // k within tile (even)
  const long long gc = bn + bn4;
  const bool ldbEven = ((ldb & 1LL) == 0);
#if !USE_TDM
  // manual A staging footprint: 1 row x 16 k
  const int arow = tid >> 1;
  const int akh  = (tid & 1) << 4;      // 0 or 16
#endif

  const int stepsPerTerm = (Kd + 31) >> 5;
  const int KdCeil = stepsPerTerm << 5;
  const int S = nTerms * stepsPerTerm;
  int buf = 0;
  int term = 0, k0 = 0;

  // running B staging pointer for (term, k0)
  const bf16_t* bp = B + (long long)bk * ldb + gc;

  v8f acc[2][2];
#pragma unroll
  for (int a = 0; a < 2; ++a)
#pragma unroll
    for (int b = 0; b < 2; ++b)
      acc[a][b] = (v8f){0.f,0.f,0.f,0.f,0.f,0.f,0.f,0.f};

#if USE_TDM
  if (wave == 0) {
    tdm_load_tile_A(A + bm * lda, (uint32_t)(uintptr_t)&As[0][0],
                    (uint32_t)Kd, (uint32_t)(M - bm), (uint32_t)lda);
  }
#endif

  for (int s = 0; s < S; ++s) {
    // ---- stage B tile transposed: Bs[n][kpair] (32 k x 64 n) ----
    if (ldbEven && (gc + 4) <= N && (k0 + bk + 2) <= Kd) {
      uint2 r0 = *(const uint2*)bp;            // n4..n4+3 at k
      uint2 r1 = *(const uint2*)(bp + ldb);    // n4..n4+3 at k+1
      Bs[(bn4 + 0) * 17 + bkp] = (r0.x & 0xffffu) | (r1.x << 16);
      Bs[(bn4 + 1) * 17 + bkp] = (r0.x >> 16)     | (r1.x & 0xffff0000u);
      Bs[(bn4 + 2) * 17 + bkp] = (r0.y & 0xffffu) | (r1.y << 16);
      Bs[(bn4 + 3) * 17 + bkp] = (r0.y >> 16)     | (r1.y & 0xffff0000u);
      if (k0 + 32 < Kd) __builtin_prefetch(bp + 32 * ldb, 0, 3);
    } else {
#pragma unroll
      for (int j = 0; j < 4; ++j) {
        bool cok = (gc + j) < N;
        bf16_t lo = (cok && (k0 + bk)     < Kd) ? bp[j]       : (bf16_t)0.f;
        bf16_t hi = (cok && (k0 + bk + 1) < Kd) ? bp[ldb + j] : (bf16_t)0.f;
        Bs[(bn4 + j) * 17 + bkp] = pack2(lo, hi);
      }
    }

    // next step coordinates (no division)
    int kNext = k0 + 32;
    int tNext = term;
    if (kNext >= KdCeil) { kNext = 0; ++tNext; }

#if USE_TDM
    // ---- A tile: wait for this step's DMA, kick off the next one ----
    if (wave == 0) {
      __builtin_amdgcn_s_wait_tensorcnt(0);
      if (tNext < nTerms) {
        const bf16_t* A2 = A + (long long)tNext * sAT + bm * lda + kNext;
        tdm_load_tile_A(A2, (uint32_t)(uintptr_t)&As[buf ^ 1][0],
                        (uint32_t)(Kd - kNext), (uint32_t)(M - bm), (uint32_t)lda);
      }
    }
#else
    // ---- manual A staging (128 rows x 32 k) ----
    {
      const bf16_t* At = A + (long long)term * sAT;
      long long gr = bm + arow;
      uint32_t* dst = &As[buf][arow * 17 + (akh >> 1)];
      const bf16_t* ap = At + gr * lda + (k0 + akh);
      if (gr < M && (k0 + akh + 16) <= Kd) {
        const uint2* ap2 = (const uint2*)ap;     // 4B-aligned (lda, k even)
        uint2 p0 = ap2[0], p1 = ap2[1], p2 = ap2[2], p3 = ap2[3];
        dst[0] = p0.x; dst[1] = p0.y;
        dst[2] = p1.x; dst[3] = p1.y;
        dst[4] = p2.x; dst[5] = p2.y;
        dst[6] = p3.x; dst[7] = p3.y;
        if (k0 + 32 < Kd) __builtin_prefetch(ap + 32, 0, 3);
      } else {
#pragma unroll
        for (int j = 0; j < 8; ++j) {
          int k = k0 + akh + 2 * j;
          bf16_t lo = (gr < M && k     < Kd) ? At[gr * lda + k]     : (bf16_t)0.f;
          bf16_t hi = (gr < M && k + 1 < Kd) ? At[gr * lda + k + 1] : (bf16_t)0.f;
          dst[j] = pack2(lo, hi);
        }
      }
    }
#endif
    __syncthreads();

    // ---- load fragments (packed per documented 16-bit A/B layout) ----
    union FragU { uint32_t u[8]; v16bf v; };
    FragU af[2], bf2[2];
    const int kb = half16 * 4;                // pair offset for K+8 half
#pragma unroll
    for (int mt = 0; mt < 2; ++mt) {
      const uint32_t* src = &As[buf][(wm * 32 + mt * 16 + rc) * 17];
#pragma unroll
      for (int j = 0; j < 4; ++j) {
        af[mt].u[j]     = src[kb + j];        // K = khalf .. khalf+7
        af[mt].u[4 + j] = src[8 + kb + j];    // K = 16+khalf .. 16+khalf+7
      }
    }
#pragma unroll
    for (int nt = 0; nt < 2; ++nt) {
      const uint32_t* src = &Bs[(wn * 32 + nt * 16 + rc) * 17];
#pragma unroll
      for (int j = 0; j < 4; ++j) {
        bf2[nt].u[j]     = src[kb + j];
        bf2[nt].u[4 + j] = src[8 + kb + j];
      }
    }

#pragma unroll
    for (int mt = 0; mt < 2; ++mt)
#pragma unroll
      for (int nt = 0; nt < 2; ++nt)
        acc[mt][nt] = __builtin_amdgcn_wmma_f32_16x16x32_bf16(
            false, af[mt].v, false, bf2[nt].v,
            (short)0, acc[mt][nt], false, false);

    __syncthreads();
    buf ^= 1;

    // advance running coordinates / B pointer
    if (tNext != term) bp = B + (long long)tNext * sBT + (long long)bk * ldb + gc;
    else               bp += 32 * ldb;
    term = tNext;
    k0   = kNext;
  }

  // ---- epilogue: alpha/beta combine, store fp32 or bf16 ----
#pragma unroll
  for (int mt = 0; mt < 2; ++mt) {
    long long rbase = bm + wm * 32 + mt * 16 + half16 * 8;
    long long rlim  = M - rbase;               // valid rows in this frag half
    if (rlim <= 0) continue;
#pragma unroll
    for (int nt = 0; nt < 2; ++nt) {
      long long gcc = bn + wn * 32 + nt * 16 + rc;
      if (gcc >= N) continue;
#pragma unroll
      for (int e = 0; e < 8; ++e) {
        if (e >= rlim) break;
        long long gr = rbase + e;
        float v = alpha * acc[mt][nt][e];
        if (Add) v += beta * (float)Add[gr * ldadd + gcc];
        long long o = gr * ldc + gcc;
        if (Cf) Cf[o] = v; else Cbf[o] = (bf16_t)v;
      }
    }
  }
}

// ---------------------------------------------------------------------------
// Elementwise helper kernels (channel counts are powers of two -> mask/shift)
// ---------------------------------------------------------------------------
__global__ void k_zero_f32(float* p, int n) {
  int i = blockIdx.x * blockDim.x + threadIdx.x;
  if (i < n) p[i] = 0.f;
}

__global__ void k_f32_to_bf16(const float* __restrict__ s, bf16_t* __restrict__ d, int n) {
  int i = blockIdx.x * blockDim.x + threadIdx.x;
  if (i < n) d[i] = (bf16_t)s[i];
}

// x: (4,8,64,768) f32 -> X0: row = n*256 + b*64 + p, F=8, bf16
__global__ void k_transpose_in(const float* __restrict__ x, bf16_t* __restrict__ out) {
  int i = blockIdx.x * blockDim.x + threadIdx.x;
  if (i >= 4 * 8 * 64 * 768) return;
  int c  = i & 7;
  int r  = i >> 3;
  int n  = r >> 8;
  int bb = r & 255;
  int b  = bb >> 6;
  int p  = bb & 63;
  out[i] = (bf16_t)x[(((b * 8 + c) * 64 + p) * 768) + n];
}

// y: (768*256, 1) f32 node-major -> out: (4,1,8,8,768) f32
__global__ void k_transpose_out(const float* __restrict__ y, float* __restrict__ out) {
  int i = blockIdx.x * blockDim.x + threadIdx.x;
  if (i >= 256 * 768) return;
  int p = i / 768;
  int n = i - p * 768;
  out[i] = y[(long long)n * 256 + p];
}

// healpix 4:1 max pool over node groups (node-major rows); fsh = log2(F)
__global__ void k_pool4(const bf16_t* __restrict__ in, bf16_t* __restrict__ out,
                        int fsh, long long total) {
  long long i = (long long)blockIdx.x * blockDim.x + threadIdx.x;
  if (i >= total) return;
  int F = 1 << fsh;
  int f = (int)(i & (F - 1));
  long long r = i >> fsh;          // (n', b)
  long long n = r >> 8;
  long long b = r & 255;
  float m = -3.0e38f;
#pragma unroll
  for (int j = 0; j < 4; ++j) {
    float v = (float)in[(((((n << 2) + j) << 8) + b) << fsh) + f];
    m = fmaxf(m, v);
  }
  out[i] = (bf16_t)m;
}

// out[(n,b)] = concat( h[(n/4,b), 0:Fh], s[(n,b), 0:Fs] ); Fh==Fs here
__global__ void k_unpool_concat(const bf16_t* __restrict__ h, int fhsh,
                                const bf16_t* __restrict__ s,
                                bf16_t* __restrict__ out, long long total) {
  long long i = (long long)blockIdx.x * blockDim.x + threadIdx.x;
  if (i >= total) return;
  int Fh = 1 << fhsh;
  int F  = Fh << 1;
  int f = (int)(i & (F - 1));
  long long r = i >> (fhsh + 1);
  long long n = r >> 8;
  long long b = r & 255;
  bf16_t v;
  if (f < Fh) v = h[((((n >> 2) << 8) + b) << fhsh) + f];
  else        v = s[(r << fhsh) + (f - Fh)];
  out[i] = v;
}

// per-channel sum / sumsq over fp32 GEMM output (F <= 256, power of two)
__global__ void k_bn_stats(const float* __restrict__ y, long long total, int fsh,
                           float* __restrict__ sum, float* __restrict__ sumsq) {
  __shared__ float ls[256], lq[256];
  int F = 1 << fsh;
  int t = threadIdx.x;
  if (t < F) { ls[t] = 0.f; lq[t] = 0.f; }
  __syncthreads();
  long long stride = (long long)gridDim.x * blockDim.x;
  for (long long i = (long long)blockIdx.x * blockDim.x + t; i < total; i += stride) {
    float v = y[i];
    int c = (int)(i & (F - 1));
    atomicAdd(&ls[c], v);
    atomicAdd(&lq[c], v * v);
  }
  __syncthreads();
  if (t < F) {
    atomicAdd(&sum[t],   ls[t]);
    atomicAdd(&sumsq[t], lq[t]);
  }
}

__global__ void k_bn_apply(const float* __restrict__ y, long long total, int fsh,
                           float invR,
                           const float* __restrict__ sum, const float* __restrict__ sumsq,
                           const float* __restrict__ g, const float* __restrict__ bta,
                           bf16_t* __restrict__ out) {
  long long i = (long long)blockIdx.x * blockDim.x + threadIdx.x;
  if (i >= total) return;
  int f = (int)(i & ((1 << fsh) - 1));
  float mu  = sum[f] * invR;
  float var = sumsq[f] * invR - mu * mu;
  float v = (y[i] - mu) * rsqrtf(var + EPSBN) * g[f] + bta[f];
  out[i] = (bf16_t)fmaxf(v, 0.f);
}

// ---------------------------------------------------------------------------
// Host orchestration
// ---------------------------------------------------------------------------
static inline unsigned cdiv(long long a, long long b) { return (unsigned)((a + b - 1) / b); }
static inline int ilog2i(int v) { int s = 0; while ((1 << s) < v) ++s; return s; }

static void launch_gemm(hipStream_t st,
                        const bf16_t* A, long long lda, long long sAT,
                        const bf16_t* B, long long ldb, long long sBT, int terms,
                        long long M, long long N, int Kd, float alpha,
                        const bf16_t* Add, long long ldadd, float beta,
                        float* Cf, bf16_t* Cbf, long long ldc) {
  dim3 g(cdiv(N, 64), cdiv(M, 128));
  k_gemm_wmma<<<g, 256, 0, st>>>(A, lda, sAT, B, ldb, sBT, terms,
                                 M, N, Kd, alpha, Add, ldadd, beta, Cf, Cbf, ldc);
}

extern "C" void kernel_launch(void* const* d_in, const int* in_sizes, int n_in,
                              void* d_out, int out_size, void* d_ws, size_t ws_size,
                              hipStream_t stream) {
  (void)in_sizes; (void)n_in; (void)out_size; (void)ws_size;

  const float* xin = (const float*)d_in[0];
  const float* Lf[4] = {(const float*)d_in[1], (const float*)d_in[2],
                        (const float*)d_in[3], (const float*)d_in[4]};
  const int Lsz[4]  = {768 * 768, 192 * 192, 48 * 48, 12 * 12};
  const long long R3 = 768LL * B2, R2 = 192LL * B2, R1 = 48LL * B2, R0 = 12LL * B2;

  // ---- workspace bump allocation (~310 MB) ----
  uint8_t* wp = (uint8_t*)d_ws;
  auto take = [&](size_t bytes) -> void* {
    void* r = wp; wp += (bytes + 255) & ~(size_t)255; return r;
  };
  bf16_t* Lb[4];
  for (int i = 0; i < 4; ++i) Lb[i] = (bf16_t*)take((size_t)Lsz[i] * 2);
  const long long Tcap = 25165824LL;                     // R3 * 128 (max Fin@768)
  bf16_t* T0 = (bf16_t*)take((size_t)4 * Tcap * 2);
  bf16_t* T1 = T0 + Tcap;
  bf16_t* T2 = T1 + Tcap;
  bf16_t* T3 = T2 + Tcap;
  bf16_t* S3 = (bf16_t*)take((size_t)R3 * 64  * 2);
  bf16_t* S2 = (bf16_t*)take((size_t)R2 * 128 * 2);
  bf16_t* S1 = (bf16_t*)take((size_t)R1 * 256 * 2);
  bf16_t* Hb = (bf16_t*)take((size_t)R2 * 64  * 2);      // covers R0*256, R1*128, R2*64
  float*  Yf = (float*)take((size_t)R3 * 64 * 4);        // largest fp32 GEMM out (e3b)
  float*  st = (float*)take(512 * 4);

  // weights -> bf16
  const int wsz[9]  = {1024, 8192, 32768, 131072, 262144, 262144, 65536, 16384, 128};
  const int widx[9] = {5, 8, 11, 14, 17, 20, 23, 26, 29};
  bf16_t* Wb[9];
  for (int i = 0; i < 9; ++i) Wb[i] = (bf16_t*)take((size_t)wsz[i] * 2);

  for (int i = 0; i < 4; ++i)
    k_f32_to_bf16<<<cdiv(Lsz[i], 256), 256, 0, stream>>>(Lf[i], Lb[i], Lsz[i]);
  for (int i = 0; i < 9; ++i)
    k_f32_to_bf16<<<cdiv(wsz[i], 256), 256, 0, stream>>>((const float*)d_in[widx[i]],
                                                         Wb[i], wsz[i]);

  // input reshape: (4,8,8,8,768) -> node-major (768,256,8) bf16 in T0
  k_transpose_in<<<cdiv(1572864, 256), 256, 0, stream>>>(xin, T0);

  // Chebyshev layer: T1..T3 via L-GEMMs (recurrence fused in epilogue),
  // then y = sum_k T_k @ W_k as a single 4-term WMMA GEMM.
  auto cheb = [&](const bf16_t* L, int Nn, int Fin, int Fout, const bf16_t* W) {
    long long cols = (long long)B2 * Fin;
    long long R = (long long)Nn * B2;
    launch_gemm(stream, L, Nn, 0, T0, cols, 0, 1, Nn, cols, Nn,
                1.f, nullptr, 0, 0.f, nullptr, T1, cols);
    launch_gemm(stream, L, Nn, 0, T1, cols, 0, 1, Nn, cols, Nn,
                2.f, T0, cols, -1.f, nullptr, T2, cols);
    launch_gemm(stream, L, Nn, 0, T2, cols, 0, 1, Nn, cols, Nn,
                2.f, T1, cols, -1.f, nullptr, T3, cols);
    launch_gemm(stream, T0, Fin, Tcap, W, Fout, (long long)Fin * Fout, 4,
                R, Fout, Fin, 1.f, nullptr, 0, 0.f, Yf, nullptr, Fout);
  };
  auto bn = [&](long long R, int F, int gi, int bi, bf16_t* out) {
    long long tot = R * F;
    int fsh = ilog2i(F);
    k_zero_f32<<<2, 256, 0, stream>>>(st, 512);
    k_bn_stats<<<512, 256, 0, stream>>>(Yf, tot, fsh, st, st + 256);
    k_bn_apply<<<cdiv(tot, 256), 256, 0, stream>>>(Yf, tot, fsh, 1.f / (float)R,
                                                   st, st + 256,
                                                   (const float*)d_in[gi],
                                                   (const float*)d_in[bi], out);
  };

  // ---- encoder ----
  cheb(Lb[0], 768, 8, 32, Wb[0]);       bn(R3, 32, 6, 7, T0);        // e3a
  cheb(Lb[0], 768, 32, 64, Wb[1]);      bn(R3, 64, 9, 10, S3);       // e3b
  k_pool4<<<cdiv(R2 * 64, 256), 256, 0, stream>>>(S3, T0, 6, R2 * 64);
  cheb(Lb[1], 192, 64, 128, Wb[2]);     bn(R2, 128, 12, 13, S2);     // e2
  k_pool4<<<cdiv(R1 * 128, 256), 256, 0, stream>>>(S2, T0, 7, R1 * 128);
  cheb(Lb[2], 48, 128, 256, Wb[3]);     bn(R1, 256, 15, 16, S1);     // e1
  k_pool4<<<cdiv(R0 * 256, 256), 256, 0, stream>>>(S1, T0, 8, R0 * 256);
  cheb(Lb[3], 12, 256, 256, Wb[4]);     bn(R0, 256, 18, 19, Hb);     // e0

  // ---- decoder ----
  k_unpool_concat<<<cdiv(R1 * 512, 256), 256, 0, stream>>>(Hb, 8, S1, T0, R1 * 512);
  cheb(Lb[2], 48, 512, 128, Wb[5]);     bn(R1, 128, 21, 22, Hb);     // d1
  k_unpool_concat<<<cdiv(R2 * 256, 256), 256, 0, stream>>>(Hb, 7, S2, T0, R2 * 256);
  cheb(Lb[1], 192, 256, 64, Wb[6]);     bn(R2, 64, 24, 25, Hb);      // d2
  k_unpool_concat<<<cdiv(R3 * 128, 256), 256, 0, stream>>>(Hb, 6, S3, T0, R3 * 128);
  cheb(Lb[0], 768, 128, 32, Wb[7]);     bn(R3, 32, 27, 28, T0);      // d3
  cheb(Lb[0], 768, 32, 1, Wb[8]);                                    // out (no bn)

  k_transpose_out<<<cdiv(256 * 768, 256), 256, 0, stream>>>(Yf, (float*)d_out);
}
